// VQVAE_42640435314819
// MI455X (gfx1250) — matrix operand order repaired and checked by reading, not compiled, not run on hardware
//
#include <hip/hip_runtime.h>
#include <hip/hip_bf16.h>

// ---------------------------------------------------------------------------
// VQ-VAE forward for MI455X (gfx1250, wave32, WMMA).
// Implicit-GEMM convs on v_wmma_f32_16x16x32_f16, f32 accumulate.
//  - block 256 thr / 8 waves (2 m-waves x 4 n-waves)
//  - NWT template: wave tile 32M x (16*NWT)N -> 2*NWT WMMA per wave per chunk
//    (NWT=4 for Cout=256 layers: 1 staged A element per lane per WMMA)
//  - weights pre-packed (cold kernel) to f16 B-fragment order; hot loop loads
//    B frags straight from global (L2-resident), zero LDS / zero addr math
//  - im2col A tile (64x32 f16) in LDS, double buffered, 1 barrier per chunk,
//    paired b32 LDS stores; input padded to Cin=4 so K%32==0 for every layer
//    (no ragged-K guard -> no exec divergence in the stage loop)
//  - inter-layer activations stored f16 (== WMMA input precision)
// ---------------------------------------------------------------------------

typedef __attribute__((ext_vector_type(16))) _Float16 v16h;
typedef __attribute__((ext_vector_type(8)))  _Float16 v8h;
typedef __attribute__((ext_vector_type(8)))  float    v8f;

#define LSTR 40  // LDS A-row stride in halves: 80B rows, 16B aligned, bank-spread

// ---- fragment loaders (contiguous-k order, same for A and B) ----------------------
// A (16x32 f16): lane L -> M=L&15 ; K run [16*(L>>4),+16) contiguous
__device__ __forceinline__ v16h load_frag_a(const _Float16* t, int lane) {
  int m  = lane & 15;
  int kb = (lane >> 4) * 16;
  v8h lo = *(const v8h*)(t + m * LSTR + kb);
  v8h hi = *(const v8h*)(t + m * LSTR + 8 + kb);
  v16h r;
#pragma unroll
  for (int i = 0; i < 8; ++i) { r[i] = lo[i]; r[8 + i] = hi[i]; }
  return r;
}
// B (32x16 f16): lane L -> N=L&15 ; K run [16*(L>>4),+16) contiguous.
// Packed layout == exactly this order: frag base + lane*16 halves.
__device__ __forceinline__ v16h load_frag_b_packed(const _Float16* frag, int lane) {
  const _Float16* p = frag + lane * 16;
  v8h lo = *(const v8h*)(p);
  v8h hi = *(const v8h*)(p + 8);
  v16h r;
#pragma unroll
  for (int i = 0; i < 8; ++i) { r[i] = lo[i]; r[8 + i] = hi[i]; }
  return r;
}

// ---- im2col input fetch (f16 activations) ---------------------------------
// MODE 0: normal conv. MODE 1: transposed conv (lhs_dilation=stride,
// effective pad = KSZ-1-pad; weight flip handled at pack time).
template <int MODE, int KSZ>
__device__ __forceinline__ _Float16 fetch_in_h(const _Float16* __restrict__ in,
                                               int b, int oy, int ox, int k,
                                               int Cin, int Hin, int Win,
                                               int stride, int pad) {
  constexpr int KHW = KSZ * KSZ;
  int ic = k / KHW; int r = k - ic * KHW;   // compile-time-constant divisors
  int ky = r / KSZ; int kx = r - ky * KSZ;
  int iy, ix;
  if (MODE == 0) {
    iy = oy * stride - pad + ky;
    ix = ox * stride - pad + kx;
  } else {
    int ty = oy + (KSZ - 1 - pad) - ky;
    int tx = ox + (KSZ - 1 - pad) - kx;
    if (ty < 0 || tx < 0) return (_Float16)0.f;
    if ((ty % stride) != 0 || (tx % stride) != 0) return (_Float16)0.f;
    iy = ty / stride; ix = tx / stride;
  }
  if ((unsigned)iy >= (unsigned)Hin || (unsigned)ix >= (unsigned)Win)
    return (_Float16)0.f;
  return in[((b * Cin + ic) * Hin + iy) * Win + ix];
}

// act: 0 none, 1 leaky_relu(0.01), 2 relu, 3 sigmoid + nan_to_num
__device__ __forceinline__ float apply_act(float v, int act) {
  if (act == 1) v = (v >= 0.f) ? v : 0.01f * v;
  else if (act == 2) v = fmaxf(v, 0.f);
  else if (act == 3) { v = 1.f / (1.f + __expf(-v)); v = (v != v) ? 0.f : v; }
  return v;
}

// ---- implicit-GEMM conv ---------------------------------------------------
// Requires: (Cin*KSZ*KSZ) % 32 == 0; M = 32*Hout*Wout % 64 == 0.
template <int MODE, int KSZ, bool OUTF32, int NWT>
__global__ __launch_bounds__(256) void conv_wmma(
    const _Float16* __restrict__ in, const _Float16* __restrict__ wpack,
    const float* __restrict__ bias, const _Float16* __restrict__ res,
    void* __restrict__ outv,
    int Cin, int Hin, int Win, int Cout, int Hout, int Wout,
    int stride, int pad, int act) {
  constexpr int KHW = KSZ * KSZ;
  __shared__ __align__(16) _Float16 Alds[2][64 * LSTR];
  __shared__ __align__(16) float    Tlds[8][16 * 17];

  const int tid  = threadIdx.x;
  const int lane = tid & 31;
  const int wave = tid >> 5;
  const int mw = wave & 1;    // 2 wave-rows (32 M each)
  const int nw = wave >> 1;   // 4 wave-cols (16*NWT N each)
  const int K   = Cin * KHW;  // multiple of 32 by construction
  const int Kc  = K >> 5;
  const int Ntiles = (Cout + 15) >> 4;
  const int HWo = Hout * Wout;
  const int mBase  = blockIdx.x * 64;
  const int ntBase = blockIdx.y * 4 * NWT;

  // hoisted output-position decomposition for the 4 staged pairs per thread
  int pb[4], poy[4], pox[4];
#pragma unroll
  for (int s = 0; s < 4; ++s) {
    int mi = (tid + s * 256) >> 4;          // 0..63
    int gm = mBase + mi;
    int b = gm / HWo; int sr = gm - b * HWo;
    pb[s] = b; poy[s] = sr / Wout; pox[s] = sr - (sr / Wout) * Wout;
  }

  auto stage = [&](int buf, int kk) {
#pragma unroll
    for (int s = 0; s < 4; ++s) {
      int t  = tid + s * 256;               // 0..1023 (pairs)
      int mi = t >> 4;
      int kp = (t & 15) * 2;                // even k within chunk
      union { _Float16 h[2]; unsigned int u; } pr;
#pragma unroll
      for (int e = 0; e < 2; ++e)
        pr.h[e] = fetch_in_h<MODE, KSZ>(in, pb[s], poy[s], pox[s], kk + kp + e,
                                        Cin, Hin, Win, stride, pad);
      *(unsigned int*)&Alds[buf][mi * LSTR + kp] = pr.u;
    }
  };

  v8f acc[2][NWT];
#pragma unroll
  for (int i = 0; i < 2; ++i)
#pragma unroll
    for (int j = 0; j < NWT; ++j)
#pragma unroll
      for (int v = 0; v < 8; ++v) acc[i][j][v] = 0.f;

  stage(0, 0);
  __syncthreads();

  for (int kc = 0; kc < Kc; ++kc) {
    const int cur = kc & 1;
    v16h bf[NWT];
#pragma unroll
    for (int j = 0; j < NWT; ++j) {
      int nt = ntBase + nw * NWT + j;
      nt = (nt < Ntiles) ? nt : (Ntiles - 1);   // clamp when Cout < block N
      bf[j] = load_frag_b_packed(wpack + (((size_t)nt * Kc + kc) << 9), lane);
    }
    v16h af[2];
#pragma unroll
    for (int i = 0; i < 2; ++i)
      af[i] = load_frag_a(&Alds[cur][(mw * 32 + i * 16) * LSTR], lane);
    if (kc + 1 < Kc) stage(cur ^ 1, (kc + 1) * 32);  // overlap next stage
#pragma unroll
    for (int i = 0; i < 2; ++i)
#pragma unroll
      for (int j = 0; j < NWT; ++j)
        acc[i][j] = __builtin_amdgcn_wmma_f32_16x16x32_f16(
            false, af[i], false, bf[j], (short)0, acc[i][j], false, false);
    __syncthreads();
  }

  // epilogue: per-subtile LDS transpose for coalesced NCHW stores
  float* T = Tlds[wave];
#pragma unroll
  for (int i = 0; i < 2; ++i) {
#pragma unroll
    for (int j = 0; j < NWT; ++j) {
      __syncthreads();
      int mh = (lane >> 4) << 3;
#pragma unroll
      for (int v = 0; v < 8; ++v) T[(v + mh) * 17 + (lane & 15)] = acc[i][j][v];
      __syncthreads();
#pragma unroll
      for (int q = 0; q < 8; ++q) {
        int nl = (lane >> 4) + 2 * q;
        int ml = lane & 15;
        int gm = mBase + mw * 32 + i * 16 + ml;
        int gn = (ntBase + nw * NWT + j) * 16 + nl;
        if (gn < Cout) {
          float v = T[ml * 17 + nl];
          if (bias) v += bias[gn];
          int b = gm / HWo; int sr = gm - b * HWo;
          int idx = (b * Cout + gn) * HWo + sr;
          if (res) v += (float)res[idx];
          v = apply_act(v, act);
          if (OUTF32) ((float*)outv)[idx] = v;
          else        ((_Float16*)outv)[idx] = (_Float16)v;
        }
      }
    }
  }
}

// ---- cold weight-pack kernel: f32 -> f16 in B-fragment order --------------
// packed[(nt*Kc + kc)*512 + lane*16 + j] = W[n = nt*16 + (lane&15),
//                                           k = kc*32 + (lane>>4)*16 + j]
// Cin may be padded (> CinReal); padded input channels pack as zero.
__global__ void pack_weights(const float* __restrict__ w, _Float16* __restrict__ out,
                             int Cin, int CinReal, int Cout, int KS, int mode,
                             int Ntiles, int Kc) {
  int gid = blockIdx.x * 256 + threadIdx.x;
  int total = Ntiles * Kc * 32;
  if (gid >= total) return;
  int lanei = gid & 31;
  int frag  = gid >> 5;
  int kc = frag % Kc;
  int nt = frag / Kc;
  int khw = KS * KS;
  int K = Cin * khw;
  int n  = nt * 16 + (lanei & 15);
  int kb = kc * 32 + (lanei >> 4) * 16;
  _Float16* dst = out + ((size_t)frag << 9) + lanei * 16;
  for (int j = 0; j < 16; ++j) {
    int k = kb + j;
    float v = 0.f;
    if (n < Cout && k < K) {
      int ic = k / khw; int r = k - ic * khw;
      int ky = r / KS;  int kx = r - ky * KS;
      if (ic < CinReal) {
        if (mode == 0)  // (Cout,CinReal,KH,KW)
          v = w[((n * CinReal + ic) * KS + ky) * KS + kx];
        else            // transposed: (CinReal,Cout,KH,KW), spatially flipped
          v = w[((ic * Cout + n) * KS + (KS - 1 - ky)) * KS + (KS - 1 - kx)];
      }
    }
    dst[j] = (_Float16)v;
  }
}

// x (B,3,128,128) f32 -> (B,4,128,128) f16, 4th channel zero (so e0 K=64)
__global__ void cvt_pad_x(const float* __restrict__ in, _Float16* __restrict__ out,
                          int HW) {
  int i = blockIdx.x * 256 + threadIdx.x;   // over 32*4*HW
  if (i >= 32 * 4 * HW) return;
  int s = i % HW; int bc = i / HW; int c = bc & 3; int b = bc >> 2;
  out[i] = (c < 3) ? (_Float16)in[(b * 3 + c) * HW + s] : (_Float16)0.f;
}

// ---- vector quantizer (f16 lat/z, f32 emb) --------------------------------
__global__ __launch_bounds__(128) void vq_kernel(
    const _Float16* __restrict__ lat, const float* __restrict__ emb,
    _Float16* __restrict__ z, float* __restrict__ loss, int HW) {
  __shared__ float Elds[128 * 64];
  __shared__ float red[128];
  int m = blockIdx.x * 128 + threadIdx.x;
  int b = m / HW; int s = m - b * HW;
  float vec[64];
#pragma unroll
  for (int c = 0; c < 64; ++c) vec[c] = (float)lat[(b * 64 + c) * HW + s];

  float bestd = 3.4e38f; int besti = 0;
  for (int p = 0; p < 4; ++p) {
    __syncthreads();
    for (int i = threadIdx.x; i < 128 * 64; i += 128)
      Elds[i] = emb[p * 8192 + i];
    __syncthreads();
    for (int e = 0; e < 128; ++e) {
      float d = 0.f;
#pragma unroll
      for (int c = 0; c < 64; ++c) {
        float t = vec[c] - Elds[e * 64 + c];
        d = fmaf(t, t, d);
      }
      if (d < bestd) { bestd = d; besti = p * 128 + e; }  // first-min == argmin
    }
  }
  float sum = 0.f;
#pragma unroll
  for (int c = 0; c < 64; ++c) {
    float q = emb[besti * 64 + c];
    float t = q - vec[c];
    sum = fmaf(t, t, sum);
    z[(b * 64 + c) * HW + s] = (_Float16)q;  // straight-through forward value
  }
  red[threadIdx.x] = sum;
  __syncthreads();
  for (int off = 64; off > 0; off >>= 1) {
    if (threadIdx.x < off) red[threadIdx.x] += red[threadIdx.x + off];
    __syncthreads();
  }
  if (threadIdx.x == 0) atomicAdd(loss, red[0]);
}

__global__ void init_loss(float* loss) { *loss = 0.f; }
// vq_loss = BETA*commit + emb_loss = 1.25 * mean((q-lat)^2)
__global__ void finalize_loss(const float* loss, float* out) {
  *out = 1.25f * (*loss) / (32768.f * 64.f);
}

// ---------------------------------------------------------------------------
extern "C" void kernel_launch(void* const* d_in, const int* in_sizes, int n_in,
                              void* d_out, int out_size, void* d_ws, size_t ws_size,
                              hipStream_t stream) {
  // d_in: 0:x 1:e0_w 2:e0_b 3:e1_w 4:e1_b 5..16:res_e(wa,wb)x6 17:e2_w 18:e2_b
  //       19:emb 20:d0_w 21:d0_b 22..33:res_d(wa,wb)x6 34:t0_w 35:t0_b 36:t1_w 37:t1_b
  auto INF = [&](int i) { return (const float*)d_in[i]; };

  char* ws = (char*)d_ws;
  const size_t MB = (size_t)1 << 20;
  _Float16* X16  = (_Float16*)(ws);             //  4 MB: x padded to Cin=4, f16
  _Float16* A    = (_Float16*)(ws + 4  * MB);   // 40 MB: e0 out / res tmp / t0 out
  _Float16* Bb   = (_Float16*)(ws + 44 * MB);   // 20 MB: running h
  _Float16* Dl   = (_Float16*)(ws + 64 * MB);   //  8 MB: lat (e2 out)
  _Float16* Zq   = (_Float16*)(ws + 72 * MB);   //  8 MB: quantized z
  _Float16* Wpk  = (_Float16*)(ws + 80 * MB);   // 24 MB: packed f16 weights
  float*    loss = (float*)   (ws + 104 * MB);

  // ---- cold pass: pack all 30 weight tensors into WMMA B-fragment order ----
  struct WDesc { int idx, Cin, CinReal, Cout, ks, mode; };
  WDesc wd[30]; int nw = 0;
  wd[nw++] = {1, 4, 3, 128, 4, 0};                      // e0 (Cin padded 3->4)
  wd[nw++] = {3, 128, 128, 256, 4, 0};                  // e1
  for (int i = 0; i < 6; ++i) {
    wd[nw++] = {5 + 2 * i, 256, 256, 256, 3, 0};        // res_e wa
    wd[nw++] = {6 + 2 * i, 256, 256, 256, 1, 0};        // res_e wb
  }
  wd[nw++] = {17, 256, 256, 64, 1, 0};                  // e2
  wd[nw++] = {20, 64, 64, 256, 3, 0};                   // d0
  for (int i = 0; i < 6; ++i) {
    wd[nw++] = {22 + 2 * i, 256, 256, 256, 3, 0};       // res_d wa
    wd[nw++] = {23 + 2 * i, 256, 256, 256, 1, 0};       // res_d wb
  }
  wd[nw++] = {34, 256, 256, 128, 4, 1};                 // t0 (transposed)
  wd[nw++] = {36, 128, 128, 3, 4, 1};                   // t1 (transposed)

  _Float16* pk[30];
  size_t off = 0;
  for (int l = 0; l < 30; ++l) {
    int K = wd[l].Cin * wd[l].ks * wd[l].ks;
    int Ntiles = (wd[l].Cout + 15) / 16;
    int Kc = (K + 31) / 32;
    pk[l] = Wpk + off;
    int total = Ntiles * Kc * 32;
    pack_weights<<<(total + 255) / 256, 256, 0, stream>>>(
        INF(wd[l].idx), pk[l], wd[l].Cin, wd[l].CinReal, wd[l].Cout, wd[l].ks,
        wd[l].mode, Ntiles, Kc);
    off += (size_t)Ntiles * Kc * 512;
  }
  cvt_pad_x<<<(32 * 4 * 16384 + 255) / 256, 256, 0, stream>>>(INF(0), X16, 16384);

  dim3 blk(256);
  auto grid_of = [&](int Hout, int Wout, int Cout, int nwt) {
    return dim3((32 * Hout * Wout) / 64, (Cout + 64 * nwt - 1) / (64 * nwt));
  };

  // ---- encoder ----
  conv_wmma<0, 4, false, 2><<<grid_of(64, 64, 128, 2), blk, 0, stream>>>(
      X16, pk[0], INF(2), nullptr, A, 4, 128, 128, 128, 64, 64, 2, 1, 1);
  conv_wmma<0, 4, false, 4><<<grid_of(32, 32, 256, 4), blk, 0, stream>>>(
      A, pk[1], INF(4), nullptr, Bb, 128, 64, 64, 256, 32, 32, 2, 1, 1);
  for (int i = 0; i < 6; ++i) {
    conv_wmma<0, 3, false, 4><<<grid_of(32, 32, 256, 4), blk, 0, stream>>>(
        Bb, pk[2 + 2 * i], nullptr, nullptr, A, 256, 32, 32, 256, 32, 32, 1, 1, 2);
    conv_wmma<0, 1, false, 4><<<grid_of(32, 32, 256, 4), blk, 0, stream>>>(
        A, pk[3 + 2 * i], nullptr, Bb, Bb, 256, 32, 32, 256, 32, 32, 1, 0,
        (i == 5) ? 1 : 0);  // trailing leaky_relu fused into last block
  }
  conv_wmma<0, 1, false, 2><<<grid_of(32, 32, 64, 2), blk, 0, stream>>>(
      Bb, pk[14], INF(18), nullptr, Dl, 256, 32, 32, 64, 32, 32, 1, 0, 1);

  // ---- vector quantizer ----
  init_loss<<<1, 1, 0, stream>>>(loss);
  vq_kernel<<<256, 128, 0, stream>>>(Dl, INF(19), Zq, loss, 1024);

  // ---- decoder ----
  conv_wmma<0, 3, false, 4><<<grid_of(32, 32, 256, 4), blk, 0, stream>>>(
      Zq, pk[15], INF(21), nullptr, Bb, 64, 32, 32, 256, 32, 32, 1, 1, 1);
  for (int i = 0; i < 6; ++i) {
    conv_wmma<0, 3, false, 4><<<grid_of(32, 32, 256, 4), blk, 0, stream>>>(
        Bb, pk[16 + 2 * i], nullptr, nullptr, A, 256, 32, 32, 256, 32, 32, 1, 1, 2);
    conv_wmma<0, 1, false, 4><<<grid_of(32, 32, 256, 4), blk, 0, stream>>>(
        A, pk[17 + 2 * i], nullptr, Bb, Bb, 256, 32, 32, 256, 32, 32, 1, 0,
        (i == 5) ? 1 : 0);
  }
  conv_wmma<1, 4, false, 2><<<grid_of(64, 64, 128, 2), blk, 0, stream>>>(
      Bb, pk[28], INF(35), nullptr, A, 256, 32, 32, 128, 64, 64, 2, 1, 1);
  conv_wmma<1, 4, true, 2><<<grid_of(128, 128, 3, 2), blk, 0, stream>>>(
      A, pk[29], INF(37), nullptr, (float*)d_out, 128, 64, 64, 3, 128, 128, 2, 1, 3);

  // vq_loss scalar appended after recon (32*3*128*128 = 1572864 floats)
  finalize_loss<<<1, 1, 0, stream>>>(loss, (float*)d_out + 1572864);
}
